// AAConv2d_7198365188192
// MI455X (gfx1250) — matrix operand order repaired and checked by reading, not compiled, not run on hardware
//
#include <hip/hip_runtime.h>
#include <stdint.h>

typedef unsigned short ushort_t;
typedef __attribute__((ext_vector_type(16))) __bf16 v16bf;
typedef __attribute__((ext_vector_type(8)))  __bf16 v8bf;
typedef __attribute__((ext_vector_type(8)))  float  v8f;
typedef int vsi4 __attribute__((vector_size(16)));

#define NN 8
#define CIN 256
#define MAP 32
#define L 1024              // MAP*MAP
#define HEADS 8
#define DKH 32
#define KTOT 2304           // 9*256
#define COTOT 1024          // conv(256)+q(256)+k(256)+v(256)
#define XPAD 34

#if defined(__has_builtin)
#if __has_builtin(__builtin_amdgcn_global_load_async_to_lds_b128)
#define HAVE_ASYNC_LDS 1
#endif
#if __has_builtin(__builtin_amdgcn_s_wait_asynccnt)
#define HAVE_WAIT_ASYNC 1
#endif
#endif

__device__ __forceinline__ ushort_t f2bf(float f) {
  uint32_t u = __float_as_uint(f);
  u = (u + 0x7fffu + ((u >> 16) & 1u)) >> 16;
  return (ushort_t)u;
}

__device__ __forceinline__ v16bf mk16(v8bf lo, v8bf hi) {
  v16bf r;
#pragma unroll
  for (int i = 0; i < 8; ++i) { r[i] = lo[i]; r[8 + i] = hi[i]; }
  return r;
}

__device__ __forceinline__ v8f wmma_bf16(v16bf a, v16bf b, v8f c) {
  return __builtin_amdgcn_wmma_f32_16x16x32_bf16(false, a, false, b, (short)0, c,
                                                 false, false);
}

__device__ __forceinline__ void wait_asynccnt0() {
#if defined(HAVE_WAIT_ASYNC)
  __builtin_amdgcn_s_wait_asynccnt(0);
#elif defined(HAVE_ASYNC_LDS)
  asm volatile("s_wait_asynccnt 0x0" ::: "memory");
#endif
}

// ---------------- pack kernels ----------------

// xcl[n][yy][xx][ci] bf16, padded 34x34 with zero border
__global__ void pack_x_kernel(const float* __restrict__ x, ushort_t* __restrict__ xcl) {
  int idx = blockIdx.x * blockDim.x + threadIdx.x;
  if (idx >= NN * XPAD * XPAD * 256) return;
  int ci = idx & 255;
  int t = idx >> 8;
  int xx = t % XPAD; t /= XPAD;
  int yy = t % XPAD; int n = t / XPAD;
  float v = 0.0f;
  if (yy >= 1 && yy <= MAP && xx >= 1 && xx <= MAP)
    v = x[((n * 256 + ci) * MAP + (yy - 1)) * MAP + (xx - 1)];
  xcl[idx] = f2bf(v);
}

// wpack[co][K] bf16, K = tap*256 + ci ; co blocks: conv,q,k,v
__global__ void pack_w_kernel(const float* __restrict__ cw, const float* __restrict__ qw,
                              const float* __restrict__ kw, const float* __restrict__ vw,
                              ushort_t* __restrict__ wpack) {
  int idx = blockIdx.x * blockDim.x + threadIdx.x;
  if (idx >= COTOT * KTOT) return;
  int K = idx % KTOT;
  int co = idx / KTOT;
  int tap = K >> 8;
  int ci = K & 255;
  int sel = co >> 8;
  int cl = co & 255;
  const float* W = (sel == 0) ? cw : (sel == 1) ? qw : (sel == 2) ? kw : vw;
  wpack[co * KTOT + K] = f2bf(W[(cl * 256 + ci) * 9 + tap]);
}

// wpack2[co][ci] (1x1 conv), wm/hm padded to 64 rows x 32
__global__ void pack_small_kernel(const float* __restrict__ aw,
                                  const float* __restrict__ wmat,
                                  const float* __restrict__ hmat,
                                  ushort_t* __restrict__ wpack2,
                                  ushort_t* __restrict__ wm, ushort_t* __restrict__ hm) {
  int idx = blockIdx.x * blockDim.x + threadIdx.x;
  if (idx < 65536) {
    wpack2[idx] = f2bf(aw[idx]);  // (256,256,1,1) flat
  } else if (idx < 65536 + 4096) {
    int j = idx - 65536;
    int which = j >> 11;          // 0 = width, 1 = height
    int r = (j & 2047) >> 5;
    int d = j & 31;
    const float* m = which ? hmat : wmat;
    float v = (r < 2 * MAP - 1) ? m[r * DKH + d] : 0.0f;
    (which ? hm : wm)[(j & 2047)] = f2bf(v);
  }
}

// ---------------- fused conv (conv/q/k/v) ----------------
// grid: 8 n * 64 ptiles, block 128 (4 waves); each wave does 16 co-tiles
// (4 groups of 4 accumulators).  3x18x256 bf16 input patch staged in LDS
// (async global->LDS when available), shared by all 4 waves.  Weight (B)
// fragments double-buffered in registers (depth 1) to overlap the global
// stream with the WMMA pipe without spilling.
__global__ void conv_qkv_kernel(const ushort_t* __restrict__ xcl,
                                const ushort_t* __restrict__ wpack,
                                const float* __restrict__ conv_b,
                                const float* __restrict__ q_b,
                                const float* __restrict__ k_b,
                                const float* __restrict__ v_b,
                                float* __restrict__ out,
                                ushort_t* __restrict__ q_ws,
                                ushort_t* __restrict__ k_ws,
                                ushort_t* __restrict__ v_ws) {
  __shared__ __attribute__((aligned(16))) ushort_t patch[3 * 18 * 256];  // 27 KB

  int blk = blockIdx.x;
  int n = blk >> 6;
  int ptile = blk & 63;
  int pbase = ptile * 16;
  int y = pbase >> 5;            // whole tile shares one row
  int x0 = pbase & 31;
  int tid = threadIdx.x;
  int wave = tid >> 5;
  int l = tid & 31;
  int lm = l & 15;
  int hi = l >> 4;

  // ---- stage input patch: rows y..y+2 (padded), cols x0..x0+17, all 256 ci
  // 3*18*32 = 1728 16-byte chunks
  for (int c = tid; c < 1728; c += 128) {
    int ky = c / 576;
    int rem = c - ky * 576;
    int xx = rem >> 5;
    int cio = (rem & 31) * 8;
    const ushort_t* g =
        xcl + (((n * XPAD + (y + ky)) * XPAD + (x0 + xx)) << 8) + cio;
    ushort_t* d = patch + ((ky * 18 + xx) << 8) + cio;
#if defined(HAVE_ASYNC_LDS)
    __builtin_amdgcn_global_load_async_to_lds_b128(
        (__attribute__((address_space(1))) vsi4*)g,
        (__attribute__((address_space(3))) vsi4*)d, 0, 0);
#else
    *(v8bf*)d = *(const v8bf*)g;
#endif
  }
  wait_asynccnt0();
  __syncthreads();

  for (int grp = 0; grp < 4; ++grp) {
    v8f acc[4] = {};
    // lane base of this wave's 4 co-tiles for this group
    const ushort_t* wb = wpack + ((wave * 16 + grp * 4) * 16 + lm) * KTOT;
    // preload B fragment for (kk=0, j=0)
    v16bf bcur = *(const v16bf*)(wb + hi * 16);
    for (int kk = 0; kk < KTOT / 32; ++kk) {
      int kb = kk * 32 + hi * 16;              // A K-base for this lane
      int tap = kb >> 8;                       // runs never cross taps
      int ci0 = kb & 255;
      int ky = tap / 3, kx = tap % 3;
      const ushort_t* abase = patch + (((ky * 18) + lm + kx) << 8) + ci0;
      v16bf a = mk16(*(const v8bf*)abase, *(const v8bf*)(abase + 8));
#pragma unroll
      for (int j = 0; j < 4; ++j) {
        // prefetch next B fragment (j+1, or j=0 of kk+1)
        int jn = (j == 3) ? 0 : (j + 1);
        int kkn = (j == 3) ? (kk + 1) : kk;
        v16bf bnext = bcur;
        if (kkn < KTOT / 32)
          bnext = *(const v16bf*)(wb + jn * 16 * KTOT + kkn * 32 + hi * 16);
        acc[j] = wmma_bf16(a, bcur, acc[j]);
        bcur = bnext;
      }
    }
#pragma unroll
    for (int j = 0; j < 4; ++j) {
      int cotile = wave * 16 + grp * 4 + j;
      int co = cotile * 16 + lm;
      int prow = pbase + hi * 8;               // 8 consecutive positions
      if (co < 256) {
        float bb = conv_b[co];
        float* op = out + (n * 512 + co) * L + prow;
#pragma unroll
        for (int v = 0; v < 8; ++v) op[v] = acc[j][v] + bb;
      } else if (co < 512) {
        int c = co - 256; float bb = q_b[c];
        int h = c >> 5, d = c & 31;
        ushort_t* qp = q_ws + (((n * HEADS + h) * L) + prow) * DKH + d;
#pragma unroll
        for (int v = 0; v < 8; ++v) qp[v * DKH] = f2bf(acc[j][v] + bb);
      } else if (co < 768) {
        int c = co - 512; float bb = k_b[c];
        int h = c >> 5, d = c & 31;
        ushort_t* kp = k_ws + (((n * HEADS + h) * L) + prow) * DKH + d;
#pragma unroll
        for (int v = 0; v < 8; ++v) kp[v * DKH] = f2bf(acc[j][v] + bb);
      } else {
        int c = co - 768; float bb = v_b[c];
        int h = c >> 5, d = c & 31;
        ushort_t* vp = v_ws + ((n * HEADS + h) * DKH + d) * L + prow;
#pragma unroll
        for (int v = 0; v < 8; ++v) vp[v] = f2bf(acc[j][v] + bb);
      }
    }
  }
}

// ---------------- attention (one wave per (n,h,qtile16)) ----------------
__global__ void attn_kernel(const ushort_t* __restrict__ q_ws,
                            const ushort_t* __restrict__ k_ws,
                            const ushort_t* __restrict__ v_ws,
                            const ushort_t* __restrict__ wm,
                            const ushort_t* __restrict__ hm,
                            ushort_t* __restrict__ attn_ws) {
  extern __shared__ char smem[];
  float*    S  = (float*)smem;                        // 16*1024 f32
  ushort_t* P  = (ushort_t*)(smem + 65536);           // 16*1024 bf16
  float*    RW = (float*)(smem + 65536 + 32768);      // 16*64
  float*    RH = RW + 16 * 64;                        // 16*64
  float*    red = RH + 16 * 64;                       // 32
  float*    rowsum = red + 32;                        // 16

  int nh = blockIdx.x >> 6;
  int qt = blockIdx.x & 63;
  int qbase = qt * 16;
  int n = nh >> 3, h = nh & 7;
  int l = threadIdx.x & 31;
  int lm = l & 15;
  int hi = l >> 4;
  int y = qbase >> 5;
  int x0 = qbase & 31;

  // Q A-fragment (reused for rel-bias GEMMs and QK^T)
  const ushort_t* qrow = q_ws + (nh * L + qbase + lm) * DKH;
  v16bf aq = mk16(*(const v8bf*)(qrow + hi * 16),
                  *(const v8bf*)(qrow + hi * 16 + 8));

  // RW = Q * width_mat^T, RH = Q * height_mat^T (N padded to 64)
#pragma unroll
  for (int rt = 0; rt < 4; ++rt) {
    int r = rt * 16 + lm;
    v16bf bw = *(const v16bf*)(wm + r * DKH + hi * 16);
    v8f cw = {};
    cw = wmma_bf16(aq, bw, cw);
    v16bf bh = *(const v16bf*)(hm + r * DKH + hi * 16);
    v8f ch = {};
    ch = wmma_bf16(aq, bh, ch);
#pragma unroll
    for (int v = 0; v < 8; ++v) {
      int rl = hi * 8 + v;
      RW[rl * 64 + r] = cw[v];
      RH[rl * 64 + r] = ch[v];
    }
  }
  __syncthreads();

  const float scale = 0.17677669529663687f;  // DKH^-0.5
  // double-buffered K fragments
  v16bf bk = *(const v16bf*)(k_ws + (nh * L + lm) * DKH + hi * 16);
  for (int kt = 0; kt < 64; ++kt) {
    v16bf bknext = bk;
    if (kt < 63)
      bknext = *(const v16bf*)(k_ws + (nh * L + (kt + 1) * 16 + lm) * DKH + hi * 16);
    v8f c = {};
    c = wmma_bf16(aq, bk, c);
    int kj = kt * 16 + lm;
    int yp = kj >> 5, xp = kj & 31;
#pragma unroll
    for (int v = 0; v < 8; ++v) {
      int rl = hi * 8 + v;
      int xq = x0 + rl;
      float bias = RW[rl * 64 + (xp - xq + 31)] + RH[rl * 64 + (yp - y + 31)];
      S[rl * 1024 + kj] = (c[v] + bias) * scale;
    }
    bk = bknext;
  }
  __syncthreads();

  // row max (2 lanes per row, 512 cols each)
  int rl = lm;
  int cb = hi * 512;
  const float* Srow = S + rl * 1024 + cb;
  float m = -1e30f;
  for (int c = 0; c < 512; ++c) m = fmaxf(m, Srow[c]);
  red[l] = m;
  __syncthreads();
  m = fmaxf(red[lm], red[lm + 16]);

  // exp + partial sum, P in bf16 (normalization deferred)
  ushort_t* Prow = P + rl * 1024 + cb;
  float s = 0.0f;
  for (int c = 0; c < 512; ++c) {
    float e = __expf(Srow[c] - m);
    s += e;
    Prow[c] = f2bf(e);
  }
  red[l] = s;
  __syncthreads();
  if (hi == 0) rowsum[lm] = red[lm] + red[lm + 16];
  __syncthreads();

  // attn = P * V  (N split: d 0..15 and 16..31), V double-buffered
  const ushort_t* vrow0 = v_ws + (nh * DKH + lm) * L;
  const ushort_t* vrow1 = v_ws + (nh * DKH + 16 + lm) * L;
  v8f c0 = {}, c1 = {};
  v16bf b0 = *(const v16bf*)(vrow0 + hi * 16);
  v16bf b1 = *(const v16bf*)(vrow1 + hi * 16);
  for (int kc = 0; kc < 32; ++kc) {
    int kb = kc * 32 + hi * 16;
    const ushort_t* pr = P + lm * 1024 + kb;
    v16bf ap = mk16(*(const v8bf*)pr, *(const v8bf*)(pr + 8));
    v16bf bn0 = b0, bn1 = b1;
    if (kc < 31) {
      int k0 = (kc + 1) * 32 + hi * 16;
      bn0 = *(const v16bf*)(vrow0 + k0);
      bn1 = *(const v16bf*)(vrow1 + k0);
    }
    c0 = wmma_bf16(ap, b0, c0);
    c1 = wmma_bf16(ap, b1, c1);
    b0 = bn0;
    b1 = bn1;
  }
#pragma unroll
  for (int v = 0; v < 8; ++v) {
    int r2 = hi * 8 + v;
    float inv = 1.0f / rowsum[r2];
    ushort_t* op = attn_ws + ((n * L) + qbase + r2) * 256 + h * 32;
    op[lm] = f2bf(c0[v] * inv);
    op[16 + lm] = f2bf(c1[v] * inv);
  }
}

// ---------------- final 1x1 conv on attention output ----------------
__global__ void attn_out_kernel(const ushort_t* __restrict__ attn_ws,
                                const ushort_t* __restrict__ wpack2,
                                const float* __restrict__ attn_b,
                                float* __restrict__ out) {
  int blk = blockIdx.x;
  int n = blk >> 6;
  int ptile = blk & 63;
  int pbase = ptile * 16;
  int tid = threadIdx.x;
  int wave = tid >> 5;
  int l = tid & 31;
  int lm = l & 15;
  int hi = l >> 4;

  for (int j = 0; j < 4; ++j) {
    int cotile = wave * 4 + j;
    int co = cotile * 16 + lm;
    v8f acc = {};
    for (int kk = 0; kk < 8; ++kk) {
      int kb = kk * 32 + hi * 16;
      const ushort_t* ar = attn_ws + (n * L + pbase + lm) * 256 + kb;
      v16bf a = mk16(*(const v8bf*)ar, *(const v8bf*)(ar + 8));
      v16bf b = *(const v16bf*)(wpack2 + co * 256 + kk * 32 + hi * 16);
      acc = wmma_bf16(a, b, acc);
    }
    float bb = attn_b[co];
    float* op = out + (n * 512 + 256 + co) * L + pbase + hi * 8;
#pragma unroll
    for (int v = 0; v < 8; ++v) op[v] = acc[v] + bb;
  }
}

// ---------------- host launch ----------------
extern "C" void kernel_launch(void* const* d_in, const int* in_sizes, int n_in,
                              void* d_out, int out_size, void* d_ws, size_t ws_size,
                              hipStream_t stream) {
  (void)in_sizes; (void)n_in; (void)out_size; (void)ws_size;
  const float* x      = (const float*)d_in[0];
  const float* conv_w = (const float*)d_in[1];
  const float* conv_b = (const float*)d_in[2];
  const float* q_w    = (const float*)d_in[3];
  const float* q_b    = (const float*)d_in[4];
  const float* k_w    = (const float*)d_in[5];
  const float* k_b    = (const float*)d_in[6];
  const float* v_w    = (const float*)d_in[7];
  const float* v_b    = (const float*)d_in[8];
  const float* attn_w = (const float*)d_in[9];
  const float* attn_b = (const float*)d_in[10];
  const float* wmat   = (const float*)d_in[11];
  const float* hmat   = (const float*)d_in[12];
  float* out = (float*)d_out;

  // workspace carve-up (bf16 == ushort bit patterns), 256B aligned blocks
  char* ws = (char*)d_ws;
  size_t off = 0;
  auto take = [&](size_t bytes) {
    char* p = ws + off;
    off += (bytes + 255) & ~(size_t)255;
    return p;
  };
  ushort_t* xcl    = (ushort_t*)take((size_t)NN * XPAD * XPAD * 256 * 2);
  ushort_t* wpack  = (ushort_t*)take((size_t)COTOT * KTOT * 2);
  ushort_t* wpack2 = (ushort_t*)take((size_t)256 * 256 * 2);
  ushort_t* wm     = (ushort_t*)take((size_t)64 * DKH * 2);
  ushort_t* hm     = (ushort_t*)take((size_t)64 * DKH * 2);
  ushort_t* q_ws   = (ushort_t*)take((size_t)NN * HEADS * L * DKH * 2);
  ushort_t* k_ws   = (ushort_t*)take((size_t)NN * HEADS * L * DKH * 2);
  ushort_t* v_ws   = (ushort_t*)take((size_t)NN * HEADS * L * DKH * 2);
  ushort_t* attnws = (ushort_t*)take((size_t)NN * L * 256 * 2);

  // packing
  {
    int tot = NN * XPAD * XPAD * 256;
    pack_x_kernel<<<(tot + 255) / 256, 256, 0, stream>>>(x, xcl);
  }
  {
    int tot = COTOT * KTOT;
    pack_w_kernel<<<(tot + 255) / 256, 256, 0, stream>>>(conv_w, q_w, k_w, v_w, wpack);
  }
  {
    int tot = 65536 + 4096;
    pack_small_kernel<<<(tot + 255) / 256, 256, 0, stream>>>(attn_w, wmat, hmat,
                                                             wpack2, wm, hm);
  }

  // fused conv + q/k/v
  conv_qkv_kernel<<<NN * 64, 128, 0, stream>>>(xcl, wpack, conv_b, q_b, k_b, v_b,
                                               out, q_ws, k_ws, v_ws);

  // attention
  {
    size_t smem = 65536 + 32768 + 4096 + 4096 + 128 + 64;  // 106,688 B
    (void)hipFuncSetAttribute((const void*)attn_kernel,
                              hipFuncAttributeMaxDynamicSharedMemorySize,
                              (int)smem);
    attn_kernel<<<64 * 64, 32, smem, stream>>>(q_ws, k_ws, v_ws, wm, hm, attnws);
  }

  // 1x1 projection of attention maps
  attn_out_kernel<<<NN * 64, 128, 0, stream>>>(attnws, wpack2, attn_b, out);
}
